// VectorQuantizerEMA_18159121727585
// MI455X (gfx1250) — matrix operand order repaired and checked by reading, not compiled, not run on hardware
//
#include <hip/hip_runtime.h>
#include <hip/hip_bf16.h>
#include <math.h>

// ---------------------------------------------------------------------------
// VQ-VAE EMA vector quantizer for MI455X (gfx1250, wave32, WMMA + TDM).
// Dominant cost: 16384x8192x256 GEMM for nearest-code search -> bf16 WMMA,
// with the codebook streamed into LDS by the Tensor Data Mover.
// ---------------------------------------------------------------------------

#define NUM_E   8192      // K codes
#define DIM     256       // D
#define NTOK    16384     // N = 16*32*32
#define BATCH   16
#define HW      1024      // 32*32
#define CHW     262144    // 256*1024
#define DECAY_F 0.99f
#define EPS_F   1e-5f

// padded LDS row: 256 bf16 (512 B) + 16 B pad = 264 ushorts (TDM pad_enable)
#define LROW    264

typedef __attribute__((ext_vector_type(16))) __bf16 v16bf_t;
typedef __attribute__((ext_vector_type(8)))  float  v8f_t;
typedef unsigned int u32x4 __attribute__((ext_vector_type(4)));
typedef int          i32x8 __attribute__((ext_vector_type(8)));
typedef int          i32x4 __attribute__((ext_vector_type(4)));

union FragU { v16bf_t v; uint4 q[2]; };

__device__ __forceinline__ unsigned short f2bf(float f) {
    unsigned int u = __float_as_uint(f);
    if ((u & 0x7fffffffu) > 0x7f800000u)          // NaN: keep quiet NaN
        return (unsigned short)((u >> 16) | 0x40u);
    return (unsigned short)((u + 0x7fffu + ((u >> 16) & 1u)) >> 16);  // RNE
}

// TDM: load a 32-row x 256-bf16 tile of the codebook into LDS, with a
// 4-DWORD pad after every 128-DWORD row (bank-conflict skew).
__device__ __forceinline__ void tdm_load_tile(const unsigned short* gsrc,
                                              unsigned int lds_off_bytes) {
    unsigned long long ga = (unsigned long long)(uintptr_t)gsrc;
    u32x4 g0;
    g0.x = 1u;                                        // count=1, no gather
    g0.y = lds_off_bytes;                             // lds_addr
    g0.z = (unsigned int)(ga & 0xffffffffu);          // global_addr[31:0]
    g0.w = (unsigned int)((ga >> 32) & 0x1ffffffu)    // global_addr[56:32]
         | (2u << 30);                                // type=2 ("image")
    i32x8 g1;
    g1[0] = (int)((1u << 16)        // data_size = 2 bytes
                | (1u << 20)        // pad_enable
                | (6u << 22)        // pad_interval: 128 DWORDs (512 B row)
                | (3u << 25));      // pad_amount: 4 DWORDs (16 B)
    g1[1] = (int)(256u << 16);      // tensor_dim0 = 256
    g1[2] = (int)(32u << 16);       // tensor_dim0[31:16]=0 | tensor_dim1=32
    g1[3] = (int)(256u << 16);      // tensor_dim1[31:16]=0 | tile_dim0=256
    g1[4] = 32;                     // tile_dim1 = 32, tile_dim2 = 0
    g1[5] = 256;                    // tensor_dim0_stride = 256
    g1[6] = 0;
    g1[7] = 0;
    i32x4 z4 = {0, 0, 0, 0};                 // groups 2/3 unused (2-D tile)
    i32x8 z8 = {0, 0, 0, 0, 0, 0, 0, 0};     // extra group (6-arg toolchain)
    __builtin_amdgcn_tensor_load_to_lds(g0, g1, z4, z4, z8, 0);
}

// --------------------------------------------------------------------------
// 0) zero the accumulator region of the workspace (counts, dw, loss)
// --------------------------------------------------------------------------
__global__ void vq_zero_kernel(float* p, int n) {
    int i = blockIdx.x * blockDim.x + threadIdx.x;
    int stride = gridDim.x * blockDim.x;
    for (; i < n; i += stride) p[i] = 0.0f;
}

// --------------------------------------------------------------------------
// 1) embedding f32 -> bf16 staging + ||e||^2 per code (one wave per row)
// --------------------------------------------------------------------------
__global__ __launch_bounds__(256) void vq_embconv_kernel(
    const float* __restrict__ emb, unsigned short* __restrict__ ebf,
    float* __restrict__ enorm)
{
    int lane = threadIdx.x & 31;
    int wave = threadIdx.x >> 5;
    int k = blockIdx.x * 8 + wave;            // 1024 blocks * 8 waves = 8192
    const float* row = emb + (size_t)k * DIM + lane * 8;
    float4 a = *(const float4*)(row);
    float4 b = *(const float4*)(row + 4);
    unsigned short t[8];
    t[0]=f2bf(a.x); t[1]=f2bf(a.y); t[2]=f2bf(a.z); t[3]=f2bf(a.w);
    t[4]=f2bf(b.x); t[5]=f2bf(b.y); t[6]=f2bf(b.z); t[7]=f2bf(b.w);
    *(uint4*)(ebf + (size_t)k * DIM + lane * 8) = *(uint4*)t;
    float s = a.x*a.x + a.y*a.y + a.z*a.z + a.w*a.w
            + b.x*b.x + b.y*b.y + b.z*b.z + b.w*b.w;
    #pragma unroll
    for (int m = 16; m > 0; m >>= 1) s += __shfl_xor(s, m, 32);
    if (lane == 0) enorm[k] = s;
}

// --------------------------------------------------------------------------
// 2) x NCHW f32 -> flat [N,D] bf16 via LDS-tiled transpose (32x32 tiles)
// --------------------------------------------------------------------------
__global__ __launch_bounds__(256) void vq_xflat_kernel(
    const float* __restrict__ x, unsigned short* __restrict__ xflat)
{
    __shared__ unsigned short tile[32][33];
    int b  = blockIdx.x >> 8;          // 16
    int dT = (blockIdx.x >> 5) & 7;    // 8 tiles of 32 channels
    int hT = blockIdx.x & 31;          // 32 tiles of 32 hw
    int tx = threadIdx.x & 31;
    int ty = threadIdx.x >> 5;         // 0..7
    const float* xb = x + (size_t)b * CHW;
    #pragma unroll
    for (int i = 0; i < 4; ++i) {
        int d = dT * 32 + ty + i * 8;
        tile[ty + i * 8][tx] = f2bf(xb[(size_t)d * HW + hT * 32 + tx]);
    }
    __syncthreads();
    #pragma unroll
    for (int i = 0; i < 4; ++i) {
        int hw = hT * 32 + ty + i * 8;
        xflat[(size_t)(b * HW + hw) * DIM + dT * 32 + tx] = tile[tx][ty + i * 8];
    }
}

// --------------------------------------------------------------------------
// 3) fused distance + argmin:  d = ||e||^2 - 2 x.e   (||x||^2 row-constant)
//    Block = 4 waves x 16 token rows. The codebook is swept in 32-column
//    tiles streamed by the TDM into double-buffered LDS; each wave runs two
//    independent WMMA accumulator chains per tile.
// --------------------------------------------------------------------------
__global__ __launch_bounds__(128) void vq_argmin_kernel(
    const unsigned short* __restrict__ xflat,
    const unsigned short* __restrict__ ebf,
    const float* __restrict__ enorm, int* __restrict__ idx)
{
    __shared__ alignas(16) unsigned short ebuf[2][32 * LROW];  // 2 x 16.5 KB

    int lane = threadIdx.x & 31;
    int wave = threadIdx.x >> 5;
    int rowBase = (blockIdx.x * 4 + wave) * 16;
    int l15  = lane & 15;
    int hiH  = lane >> 4;                // 0 or 1 (half-wave)

    // A fragments: 16x32 bf16 tiles, K=0..255 resident in registers
    FragU a[8];
    const unsigned short* xr = xflat + (size_t)(rowBase + l15) * DIM + hiH * 16;
    #pragma unroll
    for (int kk = 0; kk < 8; ++kk) {
        a[kk].q[0] = *(const uint4*)(xr + kk * 32);
        a[kk].q[1] = *(const uint4*)(xr + kk * 32 + 8);
    }

    float best[8]; int bidx[8];
    #pragma unroll
    for (int r = 0; r < 8; ++r) { best[r] = 3.4e38f; bidx[r] = 0; }

    const int k16 = hiH * 16;

    // prologue: stream tile 0
    if (wave == 0) {
        tdm_load_tile(ebf, (unsigned int)(uintptr_t)&ebuf[0][0]);
        __builtin_amdgcn_s_wait_tensorcnt((short)0);
    }
    __syncthreads();

    for (int ct = 0; ct < NUM_E / 32; ++ct) {
        int cur = ct & 1;
        if (wave == 0 && ct + 1 < NUM_E / 32)
            tdm_load_tile(ebf + (size_t)(ct + 1) * 32 * DIM,
                          (unsigned int)(uintptr_t)&ebuf[cur ^ 1][0]);

        const unsigned short* l0 = &ebuf[cur][0] + l15 * LROW + k16;
        const unsigned short* l1 = l0 + 16 * LROW;
        v8f_t acc0 = {0.f,0.f,0.f,0.f,0.f,0.f,0.f,0.f};
        v8f_t acc1 = {0.f,0.f,0.f,0.f,0.f,0.f,0.f,0.f};
        #pragma unroll
        for (int kk = 0; kk < 8; ++kk) {
            FragU b0, b1;
            b0.q[0] = *(const uint4*)(l0 + kk * 32);
            b0.q[1] = *(const uint4*)(l0 + kk * 32 + 8);
            b1.q[0] = *(const uint4*)(l1 + kk * 32);
            b1.q[1] = *(const uint4*)(l1 + kk * 32 + 8);
            acc0 = __builtin_amdgcn_wmma_f32_16x16x32_bf16(
                false, a[kk].v, false, b0.v, (short)0, acc0, false, false);
            acc1 = __builtin_amdgcn_wmma_f32_16x16x32_bf16(
                false, a[kk].v, false, b1.v, (short)0, acc1, false, false);
        }
        int   col0 = ct * 32 + l15;
        float en0  = enorm[col0];
        float en1  = enorm[col0 + 16];
        #pragma unroll
        for (int r = 0; r < 8; ++r) {
            float d0 = en0 - 2.0f * acc0[r];
            if (d0 < best[r]) { best[r] = d0; bidx[r] = col0; }
            float d1 = en1 - 2.0f * acc1[r];
            if (d1 < best[r]) { best[r] = d1; bidx[r] = col0 + 16; }
        }
        if (wave == 0) __builtin_amdgcn_s_wait_tensorcnt((short)0);
        __syncthreads();
    }

    // argmin across 16 lanes of each half-wave (row M = r + 8*hiH)
    #pragma unroll
    for (int r = 0; r < 8; ++r) {
        float bv = best[r]; int bi = bidx[r];
        #pragma unroll
        for (int m = 1; m < 16; m <<= 1) {
            float ov = __shfl_xor(bv, m, 32);
            int   oi = __shfl_xor(bi, m, 32);
            if (ov < bv || (ov == bv && oi < bi)) { bv = ov; bi = oi; }
        }
        if (l15 == 0) idx[rowBase + r + hiH * 8] = bi;
    }
}

// --------------------------------------------------------------------------
// 4) gather + straight-through output + commitment-loss partial sums
// --------------------------------------------------------------------------
__global__ __launch_bounds__(256) void vq_gather_kernel(
    const float* __restrict__ x, const float* __restrict__ emb,
    const int* __restrict__ idx, float* __restrict__ qout,
    float* __restrict__ lossAcc)
{
    __shared__ float red[256];
    int b = blockIdx.x >> 8;
    int d = blockIdx.x & 255;
    const float* xp = x    + (size_t)b * CHW + (size_t)d * HW;
    float*       qp = qout + (size_t)b * CHW + (size_t)d * HW;
    float lsum = 0.0f;
    #pragma unroll
    for (int i = 0; i < 4; ++i) {
        int hw = threadIdx.x + i * 256;
        int k  = idx[b * HW + hw];
        float q  = emb[(size_t)k * DIM + d];
        float xv = xp[hw];
        qp[hw] = xv + (q - xv);              // straight-through (matches ref fp)
        float df = xv - q;
        lsum += df * df;
    }
    red[threadIdx.x] = lsum;
    __syncthreads();
    for (int off = 128; off > 0; off >>= 1) {
        if (threadIdx.x < off) red[threadIdx.x] += red[threadIdx.x + off];
        __syncthreads();
    }
    if (threadIdx.x == 0) atomicAdd(lossAcc, red[0]);
}

// --------------------------------------------------------------------------
// 5) segment sums: counts[k] += 1, dw[k,d] += flat[n,d]  (atomic scatter)
// --------------------------------------------------------------------------
__global__ __launch_bounds__(256) void vq_scatter_kernel(
    const float* __restrict__ x, const int* __restrict__ idx,
    float* __restrict__ counts, float* __restrict__ dw)
{
    int n = blockIdx.x;
    int d = threadIdx.x;
    int k = idx[n];
    int b = n >> 10, hw = n & 1023;
    float xv = x[(size_t)b * CHW + (size_t)d * HW + hw];
    atomicAdd(&dw[(size_t)k * DIM + d], xv);
    if (d == 0) atomicAdd(&counts[k], 1.0f);
}

// --------------------------------------------------------------------------
// 6) EMA finalize: new_cluster_size, new_embed_avg, new_embedding
// --------------------------------------------------------------------------
__global__ __launch_bounds__(256) void vq_ema_kernel(
    const float* __restrict__ cs, const float* __restrict__ ea,
    const float* __restrict__ counts, const float* __restrict__ dw,
    float* __restrict__ out_emb, float* __restrict__ out_cs,
    float* __restrict__ out_ea)
{
    int k = blockIdx.x;
    int d = threadIdx.x;
    float ncs = cs[k] * DECAY_F + (1.0f - DECAY_F) * counts[k];
    size_t o = (size_t)k * DIM + d;
    float nea = ea[o] * DECAY_F + (1.0f - DECAY_F) * dw[o];
    out_ea[o]  = nea;
    out_emb[o] = nea / fmaxf(ncs, EPS_F);
    if (d == 0) out_cs[k] = ncs;
}

// --------------------------------------------------------------------------
// 7) scalars: loss, perplexity
// --------------------------------------------------------------------------
__global__ __launch_bounds__(256) void vq_scalars_kernel(
    const float* __restrict__ counts, const float* __restrict__ lossAcc,
    float* __restrict__ out_loss, float* __restrict__ out_perp)
{
    __shared__ float red[256];
    float s = 0.0f;
    for (int k = threadIdx.x; k < NUM_E; k += 256) {
        float p = counts[k] * (1.0f / (float)NTOK);
        s += p * logf(p + 1e-10f);
    }
    red[threadIdx.x] = s;
    __syncthreads();
    for (int off = 128; off > 0; off >>= 1) {
        if (threadIdx.x < off) red[threadIdx.x] += red[threadIdx.x + off];
        __syncthreads();
    }
    if (threadIdx.x == 0) {
        *out_perp = expf(-red[0]);
        *out_loss = lossAcc[0] * (0.25f / (float)(NTOK * DIM));
    }
}

// --------------------------------------------------------------------------
extern "C" void kernel_launch(void* const* d_in, const int* in_sizes, int n_in,
                              void* d_out, int out_size, void* d_ws, size_t ws_size,
                              hipStream_t stream)
{
    const float* x   = (const float*)d_in[0];   // [16,256,32,32]
    const float* emb = (const float*)d_in[1];   // [8192,256]
    const float* cs  = (const float*)d_in[2];   // [8192]
    const float* ea  = (const float*)d_in[3];   // [8192,256]

    // workspace carve-up (bytes)
    char* ws = (char*)d_ws;
    unsigned short* ebf    = (unsigned short*)(ws);                             //  4 MB
    unsigned short* xflat  = (unsigned short*)(ws + (size_t)4 * 1024 * 1024);   //  8 MB
    float*          enorm  = (float*)(ws + (size_t)12 * 1024 * 1024);           // 32 KB
    int*            idx    = (int*)  (ws + (size_t)12 * 1024 * 1024 + 32768);   // 64 KB
    float*          counts = (float*)(ws + (size_t)12 * 1024 * 1024 + 98304);   // 32 KB
    float*          dw     = (float*)(ws + (size_t)12 * 1024 * 1024 + 131072);  //  8 MB
    float*          lossAcc= (float*)(ws + (size_t)12 * 1024 * 1024 + 131072
                                          + (size_t)NUM_E * DIM * 4);

    // output carve-up (floats, tuple order)
    float* out    = (float*)d_out;
    float* o_q    = out;                         // 4,194,304
    float* o_loss = out + 4194304;               // 1
    float* o_perp = out + 4194305;               // 1
    float* o_emb  = out + 4194306;               // 2,097,152
    float* o_cs   = out + 6291458;               // 8,192
    float* o_ea   = out + 6299650;               // 2,097,152

    // 0) zero accumulators (counts + dw + loss are contiguous)
    vq_zero_kernel<<<1024, 256, 0, stream>>>(counts, NUM_E + NUM_E * DIM + 1);
    // 1) embedding -> bf16 + norms
    vq_embconv_kernel<<<NUM_E / 8, 256, 0, stream>>>(emb, ebf, enorm);
    // 2) x -> flat bf16
    vq_xflat_kernel<<<BATCH * 8 * 32, 256, 0, stream>>>(x, xflat);
    // 3) WMMA + TDM distance + argmin
    vq_argmin_kernel<<<NTOK / 64, 128, 0, stream>>>(xflat, ebf, enorm, idx);
    // 4) gather + straight-through + loss partials
    vq_gather_kernel<<<BATCH * DIM, 256, 0, stream>>>(x, emb, idx, o_q, lossAcc);
    // 5) EMA segment sums
    vq_scatter_kernel<<<NTOK, 256, 0, stream>>>(x, idx, counts, dw);
    // 6) EMA finalize
    vq_ema_kernel<<<NUM_E, 256, 0, stream>>>(cs, ea, counts, dw, o_emb, o_cs, o_ea);
    // 7) scalars
    vq_scalars_kernel<<<1, 256, 0, stream>>>(counts, lossAcc, o_loss, o_perp);
}
